// TransformerModel_91190745629273
// MI455X (gfx1250) — matrix operand order repaired and checked by reading, not compiled, not run on hardware
//
#include <hip/hip_runtime.h>

// ---------------------------------------------------------------------------
// Types for CDNA5 WMMA (wave32): v_wmma_f32_16x16x32_bf16
// ---------------------------------------------------------------------------
typedef __bf16 bf16;
typedef bf16  v16bf __attribute__((ext_vector_type(16)));
typedef bf16  v8bf  __attribute__((ext_vector_type(8)));
typedef float v8f   __attribute__((ext_vector_type(8)));

#define DIM       4096
#define N_HEAD    32
#define N_KV      8
#define HEAD_DIM  128
#define BB        16
#define SS        128
#define MAX_SEQ   1024
#define MROWS     (BB * SS)              // 2048
#define QKV_N     ((N_HEAD + 2 * N_KV) * HEAD_DIM)  // 6144
#define SCALE_ATT 0.08838834764831845f   // 1/sqrt(128)

// ---------------------------------------------------------------------------
// CDNA5 async global->LDS DMA (ASYNCcnt-tracked), per CDNA5 ISA
// LDS operand: low 32 bits of the generic pointer == LDS byte offset.
// ---------------------------------------------------------------------------
__device__ __forceinline__ void async_copy_b128(void* ldsDst, const void* gsrc) {
    unsigned l = (unsigned)(reinterpret_cast<uintptr_t>(ldsDst));
    asm volatile("global_load_async_to_lds_b128 %0, %1, off"
                 :: "v"(l), "v"(gsrc) : "memory");
}
#define WAIT_ASYNC(n) asm volatile("s_wait_asynccnt " #n ::: "memory")
#define WAIT_DS0()    asm volatile("s_wait_dscnt 0" ::: "memory")

// ---------------------------------------------------------------------------
// Fragment loaders — contiguous k order K = j + 16*hi, same for A and B (wave32)
// ---------------------------------------------------------------------------
// A 16x32 bf16: lane holds M = lane&15; element j -> K = j + 16*hi
__device__ __forceinline__ v16bf load_a_frag(const bf16* base, int ldk) {
    int lane = threadIdx.x & 31;
    int m = lane & 15, hi = lane >> 4;
    const bf16* p = base + m * ldk + 16 * hi;
    v8bf lo  = *(const v8bf*)(p);        // K = 16*hi + 0..7
    v8bf hi8 = *(const v8bf*)(p + 8);    // K = 16*hi + 8..15
    return __builtin_shufflevector(lo, hi8, 0,1,2,3,4,5,6,7,8,9,10,11,12,13,14,15);
}

// B 32x16 bf16 from tile stored [N][K] (K contiguous): element j -> K = j+16*hi
__device__ __forceinline__ v16bf load_b_frag_k(const bf16* base, int ldk) {
    int lane = threadIdx.x & 31;
    int n = lane & 15, hi = lane >> 4;
    return *(const v16bf*)(base + n * ldk + 16 * hi);   // 32 contiguous bytes
}

// B 32x16 bf16 from tile stored [K][N] (N contiguous): strided 16-bit gathers
__device__ __forceinline__ v16bf load_b_frag_n(const bf16* base, int ldn) {
    int lane = threadIdx.x & 31;
    int n = lane & 15, hi = lane >> 4;
    v16bf f;
#pragma unroll
    for (int j = 0; j < 16; ++j) f[j] = base[(j + 16 * hi) * ldn + n];
    return f;
}

__device__ __forceinline__ v8f wmma_bf16(v16bf a, v16bf b, v8f c) {
    return __builtin_amdgcn_wmma_f32_16x16x32_bf16(false, a, false, b,
                                                   (short)0, c, false, false);
}

__device__ __forceinline__ float redmax16(float v) {
#pragma unroll
    for (int m = 1; m < 16; m <<= 1) v = fmaxf(v, __shfl_xor(v, m, 32));
    return v;
}
__device__ __forceinline__ float redsum16(float v) {
#pragma unroll
    for (int m = 1; m < 16; m <<= 1) v += __shfl_xor(v, m, 32);
    return v;
}

// ---------------------------------------------------------------------------
// Kernel 1: embedding gather  X[row] = tok_emb[idx[row]]
// ---------------------------------------------------------------------------
__global__ __launch_bounds__(256) void embed_gather(const int* __restrict__ idx,
                                                    const bf16* __restrict__ emb,
                                                    bf16* __restrict__ X) {
    int row = blockIdx.x;                       // 0..2047
    int tok = idx[row];
    const uint4* src = (const uint4*)(emb + (size_t)tok * DIM);
    uint4*       dst = (uint4*)(X + (size_t)row * DIM);
#pragma unroll
    for (int i = 0; i < 2; ++i) dst[threadIdx.x + i * 256] = src[threadIdx.x + i * 256];
}

// ---------------------------------------------------------------------------
// Kernel 2: tiled bf16 WMMA GEMM, C(MxN) = A(MxK) * Bw(NxK)^T, bf16 out
// Block tile 128x128, K-step 32, async double-buffered LDS pipeline.
// 8 waves, each owns a 32x64 sub-tile.
// ---------------------------------------------------------------------------
#define G_BM 128
#define G_BN 128
#define G_BK 32
#define G_LD 48   // padded LDS pitch (multiple of 16 for 32B frag alignment)

__device__ __forceinline__ void gemm_stage(bf16* dA, bf16* dB,
                                           const bf16* Ab, const bf16* Bb,
                                           int K, int k0, int tid) {
#pragma unroll
    for (int i = 0; i < 2; ++i) {
        int idx = tid + i * 256;          // 0..511 -> 128 rows x 4 b128
        int row = idx >> 2, c8 = (idx & 3) * 8;
        async_copy_b128(&dA[row * G_LD + c8], &Ab[(size_t)row * K + k0 + c8]);
        async_copy_b128(&dB[row * G_LD + c8], &Bb[(size_t)row * K + k0 + c8]);
    }
}

__global__ __launch_bounds__(256) void gemm_bf16_nt(const bf16* __restrict__ A,
                                                    const bf16* __restrict__ Bw,
                                                    bf16* __restrict__ C,
                                                    int M, int N, int K) {
    __shared__ bf16 As[2][G_BM * G_LD];   // 2 x 12 KB
    __shared__ bf16 Bs[2][G_BN * G_LD];   // 2 x 12 KB

    int tid  = threadIdx.x;
    int wave = tid >> 5, lane = tid & 31;
    int hi = lane >> 4, ln = lane & 15;
    int wm = (wave >> 1) * 32;   // 4 row-groups of 32
    int wn = (wave & 1) * 64;    // 2 col-groups of 64

    const bf16* Ab = A  + (size_t)blockIdx.y * G_BM * K;
    const bf16* Bb = Bw + (size_t)blockIdx.x * G_BN * K;

    v8f acc[2][4];
#pragma unroll
    for (int i = 0; i < 2; ++i)
#pragma unroll
        for (int j = 0; j < 4; ++j) acc[i][j] = (v8f)0.f;

    gemm_stage(As[0], Bs[0], Ab, Bb, K, 0, tid);      // prologue: stage 0
    int buf = 0;
    for (int k0 = 0; k0 < K; k0 += G_BK) {
        if (k0 + G_BK < K) {                          // issue next stage, then
            gemm_stage(As[buf ^ 1], Bs[buf ^ 1], Ab, Bb, K, k0 + G_BK, tid);
            WAIT_ASYNC(4);                            // wait only the prev stage
        } else {
            WAIT_ASYNC(0);
        }
        __syncthreads();
        const bf16* At = As[buf];
        const bf16* Bt = Bs[buf];
        v16bf a0 = load_a_frag(&At[(wm +  0) * G_LD], G_LD);
        v16bf a1 = load_a_frag(&At[(wm + 16) * G_LD], G_LD);
#pragma unroll
        for (int j = 0; j < 4; ++j) {
            v16bf b = load_b_frag_k(&Bt[(wn + j * 16) * G_LD], G_LD);
            acc[0][j] = wmma_bf16(a0, b, acc[0][j]);
            acc[1][j] = wmma_bf16(a1, b, acc[1][j]);
        }
        __syncthreads();                              // WAR: before buf reuse
        buf ^= 1;
    }

    int m0 = blockIdx.y * G_BM + wm;
    int n0 = blockIdx.x * G_BN + wn;
#pragma unroll
    for (int i = 0; i < 2; ++i)
#pragma unroll
        for (int j = 0; j < 4; ++j)
#pragma unroll
            for (int r = 0; r < 8; ++r) {
                int row = m0 + i * 16 + r + 8 * hi;
                int col = n0 + j * 16 + ln;
                C[(size_t)row * N + col] = (bf16)acc[i][j][r];
            }
}

// ---------------------------------------------------------------------------
// Kernel 3: RoPE on Q/K (fp32 math) + V copy; reshapes into (b,h,s,d) layout
// ---------------------------------------------------------------------------
#define QPAIRS (BB * SS * N_HEAD * 64)   // 4194304
#define KPAIRS (BB * SS * N_KV * 64)     // 1048576
#define VELEMS (BB * SS * N_KV * 128)    // 2097152

__global__ __launch_bounds__(256) void rope_scatter(const bf16* __restrict__ yy,
                                                    const float* __restrict__ fc,
                                                    bf16* __restrict__ Qw,
                                                    bf16* __restrict__ Kw,
                                                    bf16* __restrict__ Vw) {
    int i = blockIdx.x * 256 + threadIdx.x;
    if (i < QPAIRS + KPAIRS) {
        int q_side = (i < QPAIRS);
        int p = q_side ? i : (i - QPAIRS);
        int d2 = p & 63;  p >>= 6;
        int nh = q_side ? N_HEAD : N_KV;
        int h  = p % nh;  p /= nh;
        int s  = p & 127; p >>= 7;
        int b  = p;
        int off = q_side ? 0 : DIM;
        const bf16* src = yy + ((size_t)(b * SS + s)) * QKV_N + off + h * HEAD_DIM + d2 * 2;
        float x0 = (float)src[0], x1 = (float)src[1];
        float c  = fc[(s * 64 + d2) * 2];
        float sn = fc[(s * 64 + d2) * 2 + 1];
        bf16* dst = (q_side ? Qw + ((size_t)(b * N_HEAD + h) * SS + s) * HEAD_DIM
                            : Kw + ((size_t)(b * N_KV  + h) * SS + s) * HEAD_DIM) + d2 * 2;
        dst[0] = (bf16)(x0 * c - x1 * sn);
        dst[1] = (bf16)(x1 * c + x0 * sn);
    } else {
        int p = i - QPAIRS - KPAIRS;
        if (p < VELEMS) {
            int d = p & 127; p >>= 7;
            int h = p & 7;   p >>= 3;
            int s = p & 127; p >>= 7;
            int b = p;
            Vw[((size_t)(b * N_KV + h) * SS + s) * HEAD_DIM + d] =
                yy[((size_t)(b * SS + s)) * QKV_N + DIM + N_KV * HEAD_DIM + h * HEAD_DIM + d];
        }
    }
}

// ---------------------------------------------------------------------------
// Kernel 4: GQA flash attention over the full 1024-entry cache.
// One block per (b, h); 8 waves x 16 query rows; KV tile = 32,
// async double-buffered K/V staging. Q buffer overlapped with K/V buffers.
// t < 128 reads roped K/V from workspace, t >= 128 reads the input caches.
// ---------------------------------------------------------------------------
#define T_TILE 32
#define LDQ 136
#define LDK 144
#define LDV 136
// shared arena (bytes): Ks0@0(9216) Ks1@9216 Vs0@18432(8704) Vs1@27136 | Ps@35840(8192)
// Qs (34816 B) overlaps the K/V region; dead after the prologue.
#define SM_TOTAL 44032

struct AttnStageArgs {
    const bf16* Kw; const bf16* Vw; const bf16* Kc; const bf16* Vc;
    size_t kwb, kcb;
};

__device__ __forceinline__ void attn_stage(bf16* Ks, bf16* Vs,
                                           const AttnStageArgs& a, int tt, int tid) {
#pragma unroll
    for (int i = 0; i < 2; ++i) {
        int idx = tid + i * 256;
        int row = idx >> 4, c8 = (idx & 15) * 8;
        int t = tt + row;
        const bf16* ksrc = (t < SS) ? a.Kw + a.kwb + (size_t)t * HEAD_DIM
                                    : a.Kc + a.kcb + (size_t)t * HEAD_DIM;
        const bf16* vsrc = (t < SS) ? a.Vw + a.kwb + (size_t)t * HEAD_DIM
                                    : a.Vc + a.kcb + (size_t)t * HEAD_DIM;
        async_copy_b128(&Ks[row * LDK + c8], &ksrc[c8]);
        async_copy_b128(&Vs[row * LDV + c8], &vsrc[c8]);
    }
}

__global__ __launch_bounds__(256) void attention_wmma(const bf16* __restrict__ Qw,
                                                      const bf16* __restrict__ Kw,
                                                      const bf16* __restrict__ Vw,
                                                      const bf16* __restrict__ Kc,
                                                      const bf16* __restrict__ Vc,
                                                      bf16* __restrict__ Ao) {
    __shared__ uint4 smem4[SM_TOTAL / 16];
    bf16* sm = (bf16*)smem4;
    bf16* Qs    = sm;                // elems 0..17407 (prologue only)
    bf16* KsB[2] = { sm,        sm + 4608  };
    bf16* VsB[2] = { sm + 9216, sm + 13568 };
    bf16* Ps    = sm + 17920;        // 8 waves x 16 x 32

    int b = blockIdx.x >> 5, h = blockIdx.x & 31;
    int kvh = h >> 2;                // repeat_interleave by 4
    int tid = threadIdx.x, wave = tid >> 5, lane = tid & 31;
    int hi = lane >> 4, ln = lane & 15;

    AttnStageArgs sa;
    sa.Kw = Kw; sa.Vw = Vw; sa.Kc = Kc; sa.Vc = Vc;
    sa.kwb = (size_t)(b * N_KV + kvh) * SS      * HEAD_DIM;
    sa.kcb = (size_t)(b * N_KV + kvh) * MAX_SEQ * HEAD_DIM;

    // ---- prologue: async-stage Q (128x128), pull this wave's A-fragments ----
    const bf16* qsrc = Qw + ((size_t)(b * N_HEAD + h) * SS) * HEAD_DIM;
#pragma unroll
    for (int i = 0; i < 8; ++i) {
        int idx = tid + i * 256;
        int row = idx >> 4, c8 = (idx & 15) * 8;
        async_copy_b128(&Qs[row * LDQ + c8], &qsrc[(size_t)row * HEAD_DIM + c8]);
    }
    WAIT_ASYNC(0);
    __syncthreads();
    v16bf qf[4];
#pragma unroll
    for (int kf = 0; kf < 4; ++kf)
        qf[kf] = load_a_frag(&Qs[(wave * 16) * LDQ + kf * 32], LDQ);
    WAIT_DS0();           // Q reads landed in VGPRs before DMA overwrites Qs
    __syncthreads();

    v8f oacc[8];
#pragma unroll
    for (int nf = 0; nf < 8; ++nf) oacc[nf] = (v8f)0.f;
    float mrow[8], lrow[8];
#pragma unroll
    for (int r = 0; r < 8; ++r) { mrow[r] = -1e30f; lrow[r] = 0.f; }

    bf16* ps = &Ps[wave * 16 * 32];

    attn_stage(KsB[0], VsB[0], sa, 0, tid);   // prologue stage tile 0
    int buf = 0;
    for (int tt = 0; tt < MAX_SEQ; tt += T_TILE) {
        if (tt + T_TILE < MAX_SEQ) {
            attn_stage(KsB[buf ^ 1], VsB[buf ^ 1], sa, tt + T_TILE, tid);
            WAIT_ASYNC(4);
        } else {
            WAIT_ASYNC(0);
        }
        __syncthreads();
        const bf16* Ks = KsB[buf];
        const bf16* Vs = VsB[buf];

        // S = Q K^T  (16 x 32 per wave), fp32 accum
        v8f s0 = (v8f)0.f, s1 = (v8f)0.f;
#pragma unroll
        for (int kf = 0; kf < 4; ++kf) {
            v16bf bk0 = load_b_frag_k(&Ks[ 0 * LDK + kf * 32], LDK);
            v16bf bk1 = load_b_frag_k(&Ks[16 * LDK + kf * 32], LDK);
            s0 = wmma_bf16(qf[kf], bk0, s0);
            s1 = wmma_bf16(qf[kf], bk1, s1);
        }

        // online softmax (row = r + 8*hi, spread across a 16-lane half)
#pragma unroll
        for (int r = 0; r < 8; ++r) {
            float v0 = s0[r] * SCALE_ATT, v1 = s1[r] * SCALE_ATT;
            float tmax = redmax16(fmaxf(v0, v1));
            float nm = fmaxf(mrow[r], tmax);
            float alpha = __expf(mrow[r] - nm);
            mrow[r] = nm;
            float p0 = __expf(v0 - nm), p1 = __expf(v1 - nm);
            lrow[r] = lrow[r] * alpha + redsum16(p0 + p1);
#pragma unroll
            for (int nf = 0; nf < 8; ++nf) oacc[nf][r] *= alpha;
            int rl = r + 8 * hi;
            ps[rl * 32 + ln]      = (bf16)p0;
            ps[rl * 32 + 16 + ln] = (bf16)p1;
        }

        // O += P V   (P: 16x32 A-frag via same-wave LDS round-trip)
        v16bf pf = load_a_frag(ps, 32);
#pragma unroll
        for (int nf = 0; nf < 8; ++nf) {
            v16bf bv = load_b_frag_n(&Vs[nf * 16], LDV);
            oacc[nf] = wmma_bf16(pf, bv, oacc[nf]);
        }
        __syncthreads();      // WAR: all waves done with buf before DMA reuse
        buf ^= 1;
    }

    // epilogue: O /= l, write (b, s, h*128+d) for the out-proj GEMM
#pragma unroll
    for (int nf = 0; nf < 8; ++nf)
#pragma unroll
        for (int r = 0; r < 8; ++r) {
            int s   = wave * 16 + r + 8 * hi;
            int col = h * HEAD_DIM + nf * 16 + ln;
            Ao[((size_t)(b * SS + s)) * DIM + col] = (bf16)(oacc[nf][r] / lrow[r]);
        }
}

// ---------------------------------------------------------------------------
// Launch: gather -> QKV GEMM -> RoPE/scatter -> flash attention -> out GEMM
// ---------------------------------------------------------------------------
extern "C" void kernel_launch(void* const* d_in, const int* in_sizes, int n_in,
                              void* d_out, int out_size, void* d_ws, size_t ws_size,
                              hipStream_t stream) {
    const int*   idx  = (const int*)d_in[0];
    const float* fc   = (const float*)d_in[1];
    // d_in[2] mask (unused), d_in[3] input_pos (int64, == arange(S))
    const bf16*  kc   = (const bf16*)d_in[4];
    const bf16*  vc   = (const bf16*)d_in[5];
    const bf16*  emb  = (const bf16*)d_in[6];
    const bf16*  wqkv = (const bf16*)d_in[7];
    const bf16*  wo   = (const bf16*)d_in[8];
    bf16* out = (bf16*)d_out;

    bf16* X  = (bf16*)d_ws;                                 // 2048 x 4096
    bf16* YY = X  + (size_t)MROWS * DIM;                    // 2048 x 6144
    bf16* Qw = YY + (size_t)MROWS * QKV_N;                  // (b,h,s,d)
    bf16* Kw = Qw + (size_t)BB * N_HEAD * SS * HEAD_DIM;    // (b,kvh,s,d)
    bf16* Vw = Kw + (size_t)BB * N_KV * SS * HEAD_DIM;
    bf16* Ao = Vw + (size_t)BB * N_KV * SS * HEAD_DIM;      // 2048 x 4096

    embed_gather<<<MROWS, 256, 0, stream>>>(idx, emb, X);

    gemm_bf16_nt<<<dim3(QKV_N / G_BN, MROWS / G_BM), 256, 0, stream>>>(
        X, wqkv, YY, MROWS, QKV_N, DIM);

    int rope_work = QPAIRS + KPAIRS + VELEMS;
    rope_scatter<<<(rope_work + 255) / 256, 256, 0, stream>>>(YY, fc, Qw, Kw, Vw);

    attention_wmma<<<BB * N_HEAD, 256, 0, stream>>>(Qw, Kw, Vw, kc, vc, Ao);

    gemm_bf16_nt<<<dim3(DIM / G_BN, MROWS / G_BM), 256, 0, stream>>>(
        Ao, wo, out, MROWS, DIM, DIM);
}
